// GraphEmbedding_25915832664871
// MI455X (gfx1250) — compile-verified, matrix operand order checked by reading, not run on hardware
//
#include <hip/hip_runtime.h>
#include <hip/hip_bf16.h>

typedef __bf16 bf16_t;
typedef __attribute__((ext_vector_type(16))) __bf16 v16bf;
typedef __attribute__((ext_vector_type(8)))  float  v8f;

// ---------------------------------------------------------------------------
// Generic GEMM: out[M,64] = act(A[M,Kpad] @ W[Kpad,64] + bias)
// A: bf16 row-major, stride Kpad (Kpad % 32 == 0, rows padded with zeros)
// Wp: bf16 pre-packed per WMMA B-layout: [Kpad/32][4 ntiles][32 lanes][16 vals]
// The packed weights (<= 32 KB) are staged into LDS once per workgroup with
// gfx1250 async load-to-LDS (ASYNCcnt), then every wave reads B tiles from
// LDS (ds_load_b128) while streaming A from global.
// Each wave computes a 16x64 tile via 4x v_wmma_f32_16x16x32_bf16 per K-step.
// ---------------------------------------------------------------------------
__global__ __launch_bounds__(256) void k_gemm_n64(
    const bf16_t* __restrict__ A, const bf16_t* __restrict__ Wp,
    const float* __restrict__ bias, float* __restrict__ outF,
    bf16_t* __restrict__ outB, int M, int Kpad, int relu)
{
  // Max layer: Kpad=256 -> Kpad*4 = 1024 v16bf = 32 KB.
  __shared__ v16bf lds_w[1024];

  // ---- stage packed W into LDS via async load-to-LDS (16B per lane-op) ----
  {
    const int n4 = Kpad * 8;                 // number of 16-byte chunks
    const unsigned lbase = (unsigned)(uintptr_t)(void*)lds_w; // LDS offset (addr[31:0])
    const char* gsrc = (const char*)Wp;
    for (int i = threadIdx.x; i < n4; i += 256) {   // uniform trip count (n4 % 256 == 0)
      unsigned lo = lbase + (unsigned)i * 16u;
      const void* ga = gsrc + (size_t)i * 16;
      asm volatile("global_load_async_to_lds_b128 %0, %1, off"
                   :: "v"(lo), "v"(ga) : "memory");
    }
    asm volatile("s_wait_asynccnt 0x0" ::: "memory");
    __syncthreads();
  }

  const int wave = threadIdx.x >> 5;
  const int lane = threadIdx.x & 31;
  const int rowBase = (blockIdx.x * 8 + wave) * 16;
  if (rowBase >= M) return;          // whole-wave uniform exit (M % 16 == 0)
  const int l15 = lane & 15;
  const int hi8 = (lane >> 4) * 8;   // 0 or 8: K sub-block / row offset

  v8f acc[4] = {v8f{}, v8f{}, v8f{}, v8f{}};

  const bf16_t* arow = A + (size_t)(rowBase + l15) * Kpad;
  const int nkb = Kpad >> 5;

  for (int kb = 0; kb < nkb; ++kb) {
    const int k0 = kb << 5;
    // Prefetch the A stream two K-steps ahead (uniform condition).
    if (k0 + 96 <= Kpad) __builtin_prefetch(arow + k0 + 64, 0, 0);

    // A 16x32 bf16 tile: lane holds K = {k0+hi8..+7, k0+16+hi8..+7} of row l15
    union { v16bf v; uint4 u[2]; } ua;
    ua.u[0] = *(const uint4*)(arow + k0 + hi8);
    ua.u[1] = *(const uint4*)(arow + k0 + 16 + hi8);
    const v16bf a = ua.v;

    const int wb = kb * 128 + lane;          // (kb*4 + nt)*32 + lane, nt stride 32
    const v16bf b0 = lds_w[wb];
    const v16bf b1 = lds_w[wb + 32];
    const v16bf b2 = lds_w[wb + 64];
    const v16bf b3 = lds_w[wb + 96];

    acc[0] = __builtin_amdgcn_wmma_f32_16x16x32_bf16(false, a, false, b0, (short)0, acc[0], false, false);
    acc[1] = __builtin_amdgcn_wmma_f32_16x16x32_bf16(false, a, false, b1, (short)0, acc[1], false, false);
    acc[2] = __builtin_amdgcn_wmma_f32_16x16x32_bf16(false, a, false, b2, (short)0, acc[2], false, false);
    acc[3] = __builtin_amdgcn_wmma_f32_16x16x32_bf16(false, a, false, b3, (short)0, acc[3], false, false);
  }

  // C layout: VGPR r -> row rowBase + r + (lane>=16 ? 8 : 0); col = 16*nt + l15
#pragma unroll
  for (int nt = 0; nt < 4; ++nt) {
    const int col = nt * 16 + l15;
    const float b = bias[col];
#pragma unroll
    for (int r = 0; r < 8; ++r) {
      float v = acc[nt][r] + b;
      if (relu) v = v > 0.f ? v : 0.f;
      const size_t o = (size_t)(rowBase + r + hi8) * 64 + col;
      if (outF) outF[o] = v;
      if (outB) outB[o] = (bf16_t)v;
    }
  }
}

// Pack W[Kreal,64] f32 -> bf16 WMMA-B layout with zero K padding.
__global__ void k_pack_w(const float* __restrict__ W, bf16_t* __restrict__ Wp,
                         int Kreal, int Kpad)
{
  int idx = blockIdx.x * 256 + threadIdx.x;
  int total = Kpad * 64;
  if (idx >= total) return;
  int i    = idx & 15;
  int lane = (idx >> 4) & 31;
  int nt   = (idx >> 9) & 3;
  int kb   = idx >> 11;
  int v = i >> 1, lo = i & 1;
  int hi8 = (lane >> 4) * 8;
  int kin = (v < 4) ? (2 * v + hi8) : (16 + 2 * (v - 4) + hi8);
  int k = kb * 32 + kin + lo;
  int n = nt * 16 + (lane & 15);
  float val = (k < Kreal) ? W[(size_t)k * 64 + n] : 0.f;
  Wp[idx] = (bf16_t)val;
}

// ------------------------- small utility kernels ---------------------------
__global__ void k_zero(float* __restrict__ p, long long n) {
  long long t = blockIdx.x * 256ll + threadIdx.x;
  if (t < n) p[t] = 0.f;
}

__global__ void k_eb(const int* __restrict__ src, const int* __restrict__ bat,
                     int* __restrict__ eb, long long E) {
  long long t = blockIdx.x * 256ll + threadIdx.x;
  if (t < E) eb[t] = bat[src[t]];
}

// pad f32 [M,16] -> bf16 [M,32] (zeros in cols 16..31)
__global__ void k_pad16(const float* __restrict__ s, bf16_t* __restrict__ d,
                        long long M) {
  long long t = blockIdx.x * 256ll + threadIdx.x;
  if (t >= M * 32) return;
  long long i = t >> 5; int c = (int)(t & 31);
  d[t] = (bf16_t)(c < 16 ? s[i * 16 + c] : 0.f);
}

__global__ void k_scatter_add(const bf16_t* __restrict__ v, const int* __restrict__ idx,
                              float* __restrict__ acc, float* __restrict__ cnt,
                              long long rows) {
  long long t = blockIdx.x * 256ll + threadIdx.x;
  if (t >= rows * 64) return;
  long long r = t >> 6; int c = (int)(t & 63);
  int g = idx[r];
  atomicAdd(&acc[(size_t)g * 64 + c], (float)v[t]);
  if (c == 0) atomicAdd(&cnt[g], 1.f);
}

__global__ void k_scatter_add_prod(const bf16_t* __restrict__ a, const bf16_t* __restrict__ b,
                                   const int* __restrict__ idx, float* __restrict__ acc,
                                   float* __restrict__ cnt, long long rows) {
  long long t = blockIdx.x * 256ll + threadIdx.x;
  if (t >= rows * 64) return;
  long long r = t >> 6; int c = (int)(t & 63);
  int g = idx[r];
  atomicAdd(&acc[(size_t)g * 64 + c], (float)a[t] * (float)b[t]);
  if (c == 0) atomicAdd(&cnt[g], 1.f);
}

__global__ void k_finalize(float* __restrict__ acc, const float* __restrict__ cnt,
                           long long n) {
  long long t = blockIdx.x * 256ll + threadIdx.x;
  if (t >= n * 64) return;
  long long i = t >> 6;
  acc[t] /= fmaxf(cnt[i], 1.f);
}

__global__ void k_diff(const float* __restrict__ a, const float* __restrict__ b,
                       bf16_t* __restrict__ o, long long n) {
  long long t = blockIdx.x * 256ll + threadIdx.x;
  if (t < n) o[t] = (bf16_t)(a[t] - b[t]);
}

// ------------------------- concat builders (bf16) --------------------------
// rec_e input [E,256]: [xc[dst]-xc[src] (80) | ec (80) | uc[eb] (80) | pad16]
__global__ void k_build_edge_rec(const float* __restrict__ x, const bf16_t* __restrict__ xh,
    const float* __restrict__ e, const bf16_t* __restrict__ eh,
    const float* __restrict__ u, const bf16_t* __restrict__ uh,
    const int* __restrict__ src, const int* __restrict__ dst,
    const int* __restrict__ eb, bf16_t* __restrict__ A, long long E)
{
  long long t = blockIdx.x * 256ll + threadIdx.x;
  if (t >= E * 256) return;
  long long edge = t >> 8; int c = (int)(t & 255);
  float v = 0.f;
  if (c < 80) {
    int s = src[edge], d = dst[edge];
    float vd = (c < 16) ? x[(size_t)d * 16 + c] : (float)xh[(size_t)d * 64 + (c - 16)];
    float vs = (c < 16) ? x[(size_t)s * 16 + c] : (float)xh[(size_t)s * 64 + (c - 16)];
    v = vd - vs;
  } else if (c < 160) {
    int c2 = c - 80;
    v = (c2 < 16) ? e[(size_t)edge * 16 + c2] : (float)eh[(size_t)edge * 64 + (c2 - 16)];
  } else if (c < 240) {
    int c2 = c - 160; int g = eb[edge];
    v = (c2 < 16) ? u[(size_t)g * 16 + c2] : (float)uh[(size_t)g * 64 + (c2 - 16)];
  }
  A[t] = (bf16_t)v;
}

// rec_x input [N,224]: [xc (80) | mean_e (64) | uc[bat] (80)]
__global__ void k_build_node_rec(const float* __restrict__ x, const bf16_t* __restrict__ xh,
    const float* __restrict__ nmean, const float* __restrict__ u,
    const bf16_t* __restrict__ uh, const int* __restrict__ bat,
    bf16_t* __restrict__ A, long long N)
{
  long long t = blockIdx.x * 256ll + threadIdx.x;
  if (t >= N * 224) return;
  long long i = t / 224; int c = (int)(t % 224);
  float v;
  if (c < 16)       v = x[i * 16 + c];
  else if (c < 80)  v = (float)xh[i * 64 + (c - 16)];
  else if (c < 144) v = nmean[i * 64 + (c - 80)];
  else {
    int b = bat[i];
    if (c < 160) v = u[(size_t)b * 16 + (c - 144)];
    else         v = (float)uh[(size_t)b * 64 + (c - 160)];
  }
  A[t] = (bf16_t)v;
}

// rec_u input [G,224]: [mean_x (64) | mean_e (64) | uc (80) | pad16]
__global__ void k_build_u_rec(const float* __restrict__ g1, const float* __restrict__ g2,
    const float* __restrict__ u, const bf16_t* __restrict__ uh,
    bf16_t* __restrict__ A, long long G)
{
  long long t = blockIdx.x * 256ll + threadIdx.x;
  if (t >= G * 224) return;
  long long i = t / 224; int c = (int)(t % 224);
  float v = 0.f;
  if (c < 64)       v = g1[i * 64 + c];
  else if (c < 128) v = g2[i * 64 + (c - 64)];
  else if (c < 144) v = u[i * 16 + (c - 128)];
  else if (c < 208) v = (float)uh[i * 64 + (c - 144)];
  A[t] = (bf16_t)v;
}

// att_e input [E,192]: [xh[dst]-xh[src] | eh | uh[eb]]
__global__ void k_build_att_edge(const bf16_t* __restrict__ xh, const bf16_t* __restrict__ eh,
    const bf16_t* __restrict__ uh, const int* __restrict__ src,
    const int* __restrict__ dst, const int* __restrict__ eb,
    bf16_t* __restrict__ A, long long E)
{
  long long t = blockIdx.x * 256ll + threadIdx.x;
  if (t >= E * 192) return;
  long long edge = t / 192; int c = (int)(t % 192);
  float v;
  if (c < 64) {
    int s = src[edge], d = dst[edge];
    v = (float)xh[(size_t)d * 64 + c] - (float)xh[(size_t)s * 64 + c];
  } else if (c < 128) {
    v = (float)eh[(size_t)edge * 64 + (c - 64)];
  } else {
    int g = eb[edge];
    v = (float)uh[(size_t)g * 64 + (c - 128)];
  }
  A[t] = (bf16_t)v;
}

// att_x input [N,192]: [xh | mean_e_a | uh[bat]]
__global__ void k_build_att_node(const bf16_t* __restrict__ xh, const float* __restrict__ nmean,
    const bf16_t* __restrict__ uh, const int* __restrict__ bat,
    bf16_t* __restrict__ A, long long N)
{
  long long t = blockIdx.x * 256ll + threadIdx.x;
  if (t >= N * 192) return;
  long long i = t / 192; int c = (int)(t % 192);
  float v;
  if (c < 64)       v = (float)xh[i * 64 + c];
  else if (c < 128) v = nmean[i * 64 + (c - 64)];
  else              v = (float)uh[(size_t)bat[i] * 64 + (c - 128)];
  A[t] = (bf16_t)v;
}

// att_u input [G,192]: [g1 | g2 | uh]
__global__ void k_build_att_u(const float* __restrict__ g1, const float* __restrict__ g2,
    const bf16_t* __restrict__ uh, bf16_t* __restrict__ A, long long G)
{
  long long t = blockIdx.x * 256ll + threadIdx.x;
  if (t >= G * 192) return;
  long long i = t / 192; int c = (int)(t % 192);
  float v;
  if (c < 64)       v = g1[i * 64 + c];
  else if (c < 128) v = g2[i * 64 + (c - 64)];
  else              v = (float)uh[i * 64 + (c - 128)];
  A[t] = (bf16_t)v;
}

// agg input [G,192]: [mean(xw) | mean(ew) | u_a*u_h]
__global__ void k_build_agg(const float* __restrict__ g1, const float* __restrict__ g2,
    const bf16_t* __restrict__ ua, const bf16_t* __restrict__ uh,
    bf16_t* __restrict__ A, long long G)
{
  long long t = blockIdx.x * 256ll + threadIdx.x;
  if (t >= G * 192) return;
  long long i = t / 192; int c = (int)(t % 192);
  float v;
  if (c < 64)       v = g1[i * 64 + c];
  else if (c < 128) v = g2[i * 64 + (c - 64)];
  else              v = (float)ua[i * 64 + (c - 128)] * (float)uh[i * 64 + (c - 128)];
  A[t] = (bf16_t)v;
}

// ---------------------------------------------------------------------------
extern "C" void kernel_launch(void* const* d_in, const int* in_sizes, int n_in,
                              void* d_out, int out_size, void* d_ws, size_t ws_size,
                              hipStream_t stream)
{
  (void)in_sizes; (void)n_in; (void)out_size; (void)ws_size;
  const long long NN = 50000, NE = 800000, NG = 128;

  const float* X[2]  = {(const float*)d_in[0], (const float*)d_in[3]};
  const float* Ef[2] = {(const float*)d_in[1], (const float*)d_in[4]};
  const float* Uf[2] = {(const float*)d_in[2], (const float*)d_in[5]};
  const int* EIDX[2] = {(const int*)d_in[72], (const int*)d_in[74]};
  const int* BAT[2]  = {(const int*)d_in[73], (const int*)d_in[75]};

  auto Wf = [&](int m, int l) { return (const float*)d_in[6 + m * 6 + l * 2]; };
  auto Bf = [&](int m, int l) { return (const float*)d_in[6 + m * 6 + l * 2 + 1]; };

  // mlp order: 0 enc_e,1 enc_x,2 enc_u,3 rec_e,4 rec_x,5 rec_u,
  //            6 att_e,7 att_x,8 att_u,9 agg_u,10 final
  static const int KIN0[11]  = {16, 16, 16, 240, 224, 208, 192, 192, 192, 192, 64};
  static const int KPAD0[11] = {32, 32, 32, 256, 224, 224, 192, 192, 192, 192, 64};

  char* base = (char*)d_ws; size_t off = 0;
  auto alloc = [&](size_t bytes) -> void* {
    void* p = base + off;
    off += (bytes + 255) & ~(size_t)255;
    return p;
  };

  bf16_t* Wp[33];
  for (int m = 0; m < 11; ++m)
    for (int l = 0; l < 3; ++l) {
      int kp = l ? 64 : KPAD0[m];
      Wp[m * 3 + l] = (bf16_t*)alloc((size_t)kp * 64 * 2);
    }
  bf16_t* Aedge = (bf16_t*)alloc((size_t)NE * 256 * 2);
  bf16_t* EB0 = (bf16_t*)alloc((size_t)NE * 64 * 2);
  bf16_t* EB1 = (bf16_t*)alloc((size_t)NE * 64 * 2);
  bf16_t* EB2 = (bf16_t*)alloc((size_t)NE * 64 * 2);
  bf16_t* Anode = (bf16_t*)alloc((size_t)NN * 256 * 2);
  bf16_t* XB0 = (bf16_t*)alloc((size_t)NN * 64 * 2);
  bf16_t* XB1 = (bf16_t*)alloc((size_t)NN * 64 * 2);
  bf16_t* XB2 = (bf16_t*)alloc((size_t)NN * 64 * 2);
  bf16_t* Au  = (bf16_t*)alloc((size_t)NG * 256 * 2);
  bf16_t* UB0 = (bf16_t*)alloc((size_t)NG * 64 * 2);
  bf16_t* UB1 = (bf16_t*)alloc((size_t)NG * 64 * 2);
  bf16_t* UB2 = (bf16_t*)alloc((size_t)NG * 64 * 2);
  float* nacc = (float*)alloc((size_t)NN * 64 * 4);
  float* ncnt = (float*)alloc((size_t)NN * 4);
  float* g1 = (float*)alloc((size_t)NG * 64 * 4);
  float* c1 = (float*)alloc((size_t)NG * 4);
  float* g2 = (float*)alloc((size_t)NG * 64 * 4);
  float* c2 = (float*)alloc((size_t)NG * 4);
  int*   EBI = (int*)alloc((size_t)NE * 4);
  float* uembA = (float*)alloc((size_t)NG * 64 * 4);
  float* uembB = (float*)alloc((size_t)NG * 64 * 4);
  bf16_t* Adiff = (bf16_t*)alloc((size_t)NG * 64 * 2);
  float* uemb[2] = {uembA, uembB};

  auto GB = [](long long n) { return (unsigned)((n + 255) / 256); };

  // pack all MLP weights to bf16 WMMA-B layout (tiny, once per launch)
  for (int m = 0; m < 11; ++m)
    for (int l = 0; l < 3; ++l) {
      int kr = l ? 64 : KIN0[m];
      int kp = l ? 64 : KPAD0[m];
      k_pack_w<<<GB((long long)kp * 64), 256, 0, stream>>>(Wf(m, l), Wp[m * 3 + l], kr, kp);
    }

  auto gemm = [&](const bf16_t* A, int Kpad, int m, int l, long long M,
                  float* oF, bf16_t* oB, int relu) {
    k_gemm_n64<<<dim3((unsigned)((M + 127) / 128)), 256, 0, stream>>>(
        A, Wp[m * 3 + l], Bf(m, l), oF, oB, (int)M, Kpad, relu);
  };
  // 3-layer MLP: A --relu--> s1 --relu--> s2 --linear--> out (oB may alias s1)
  auto mlp3 = [&](const bf16_t* A, int Kpad, int m, long long M,
                  bf16_t* s1, bf16_t* s2, bf16_t* oB, float* oF) {
    gemm(A, Kpad, m, 0, M, nullptr, s1, 1);
    gemm(s1, 64, m, 1, M, nullptr, s2, 1);
    gemm(s2, 64, m, 2, M, oF, oB, 0);
  };

  for (int s = 0; s < 2; ++s) {
    const int* src = EIDX[s];
    const int* dst = EIDX[s] + NE;
    const int* bat = BAT[s];
    k_eb<<<GB(NE), 256, 0, stream>>>(src, bat, EBI, NE);

    // ---- encoders ----
    k_pad16<<<GB(NN * 32), 256, 0, stream>>>(X[s], Anode, NN);
    mlp3(Anode, 32, 1, NN, XB1, XB2, XB0, nullptr);              // x_h = XB0
    k_pad16<<<GB(NE * 32), 256, 0, stream>>>(Ef[s], Aedge, NE);
    mlp3(Aedge, 32, 0, NE, EB1, EB2, EB0, nullptr);              // e_h = EB0
    k_pad16<<<GB(NG * 32), 256, 0, stream>>>(Uf[s], Au, NG);
    mlp3(Au, 32, 2, NG, UB1, UB2, UB0, nullptr);                 // u_h = UB0

    // ---- recurrent rounds ----
    for (int r = 0; r < 3; ++r) {
      k_build_edge_rec<<<GB(NE * 256), 256, 0, stream>>>(
          X[s], XB0, Ef[s], EB0, Uf[s], UB0, src, dst, EBI, Aedge, NE);
      mlp3(Aedge, 256, 3, NE, EB1, EB2, EB0, nullptr);           // e_new = EB0

      k_zero<<<GB(NN * 64), 256, 0, stream>>>(nacc, NN * 64);
      k_zero<<<GB(NN), 256, 0, stream>>>(ncnt, NN);
      k_scatter_add<<<GB(NE * 64), 256, 0, stream>>>(EB0, dst, nacc, ncnt, NE);
      k_finalize<<<GB(NN * 64), 256, 0, stream>>>(nacc, ncnt, NN);

      k_build_node_rec<<<GB(NN * 224), 256, 0, stream>>>(
          X[s], XB0, nacc, Uf[s], UB0, bat, Anode, NN);
      mlp3(Anode, 224, 4, NN, XB1, XB2, XB0, nullptr);           // x_new = XB0

      k_zero<<<GB(NG * 64), 256, 0, stream>>>(g1, NG * 64);
      k_zero<<<GB(NG), 256, 0, stream>>>(c1, NG);
      k_zero<<<GB(NG * 64), 256, 0, stream>>>(g2, NG * 64);
      k_zero<<<GB(NG), 256, 0, stream>>>(c2, NG);
      k_scatter_add<<<GB(NN * 64), 256, 0, stream>>>(XB0, bat, g1, c1, NN);
      k_scatter_add<<<GB(NE * 64), 256, 0, stream>>>(EB0, EBI, g2, c2, NE);
      k_finalize<<<GB(NG * 64), 256, 0, stream>>>(g1, c1, NG);
      k_finalize<<<GB(NG * 64), 256, 0, stream>>>(g2, c2, NG);

      k_build_u_rec<<<GB(NG * 224), 256, 0, stream>>>(g1, g2, Uf[s], UB0, Au, NG);
      mlp3(Au, 224, 5, NG, UB1, UB2, UB0, nullptr);              // u_new = UB0
    }

    // ---- attention MetaLayer ----
    k_build_att_edge<<<GB(NE * 192), 256, 0, stream>>>(
        XB0, EB0, UB0, src, dst, EBI, Aedge, NE);
    mlp3(Aedge, 192, 6, NE, EB1, EB2, EB1, nullptr);             // e_a = EB1 (eh intact)

    k_zero<<<GB(NN * 64), 256, 0, stream>>>(nacc, NN * 64);
    k_zero<<<GB(NN), 256, 0, stream>>>(ncnt, NN);
    k_scatter_add<<<GB(NE * 64), 256, 0, stream>>>(EB1, dst, nacc, ncnt, NE);
    k_finalize<<<GB(NN * 64), 256, 0, stream>>>(nacc, ncnt, NN);

    k_build_att_node<<<GB(NN * 192), 256, 0, stream>>>(XB0, nacc, UB0, bat, Anode, NN);
    mlp3(Anode, 192, 7, NN, XB1, XB2, XB1, nullptr);             // x_a = XB1 (xh intact)

    k_zero<<<GB(NG * 64), 256, 0, stream>>>(g1, NG * 64);
    k_zero<<<GB(NG), 256, 0, stream>>>(c1, NG);
    k_zero<<<GB(NG * 64), 256, 0, stream>>>(g2, NG * 64);
    k_zero<<<GB(NG), 256, 0, stream>>>(c2, NG);
    k_scatter_add<<<GB(NN * 64), 256, 0, stream>>>(XB1, bat, g1, c1, NN);
    k_scatter_add<<<GB(NE * 64), 256, 0, stream>>>(EB1, EBI, g2, c2, NE);
    k_finalize<<<GB(NG * 64), 256, 0, stream>>>(g1, c1, NG);
    k_finalize<<<GB(NG * 64), 256, 0, stream>>>(g2, c2, NG);

    k_build_att_u<<<GB(NG * 192), 256, 0, stream>>>(g1, g2, UB0, Au, NG);
    mlp3(Au, 192, 8, NG, UB1, UB2, UB1, nullptr);                // u_a = UB1 (uh intact)

    // ---- aggregator (GlobalModel) ----
    k_zero<<<GB(NG * 64), 256, 0, stream>>>(g1, NG * 64);
    k_zero<<<GB(NG), 256, 0, stream>>>(c1, NG);
    k_zero<<<GB(NG * 64), 256, 0, stream>>>(g2, NG * 64);
    k_zero<<<GB(NG), 256, 0, stream>>>(c2, NG);
    k_scatter_add_prod<<<GB(NN * 64), 256, 0, stream>>>(XB1, XB0, bat, g1, c1, NN);
    k_scatter_add_prod<<<GB(NE * 64), 256, 0, stream>>>(EB1, EB0, EBI, g2, c2, NE);
    k_finalize<<<GB(NG * 64), 256, 0, stream>>>(g1, c1, NG);
    k_finalize<<<GB(NG * 64), 256, 0, stream>>>(g2, c2, NG);

    k_build_agg<<<GB(NG * 192), 256, 0, stream>>>(g1, g2, UB1, UB0, Au, NG);
    gemm(Au, 192, 9, 0, NG, nullptr, UB0, 1);                    // uh dead now
    gemm(UB0, 64, 9, 1, NG, nullptr, UB2, 1);
    gemm(UB2, 64, 9, 2, NG, uemb[s], nullptr, 0);                // graph embedding f32
  }

  // ---- final MLP on (u1e - u2e) ----
  k_diff<<<GB(NG * 64), 256, 0, stream>>>(uemb[0], uemb[1], Adiff, NG * 64);
  gemm(Adiff, 64, 10, 0, NG, nullptr, UB0, 1);
  gemm(UB0, 64, 10, 1, NG, nullptr, UB1, 1);
  gemm(UB1, 64, 10, 2, NG, (float*)d_out, nullptr, 0);
}
